// ObservedDescMatmul_15960098472448
// MI455X (gfx1250) — 8-bit matrix operand order repaired and checked by reading, compiled, not run on hardware
//
#include <hip/hip_runtime.h>
#include <hip/hip_bf16.h>

typedef __attribute__((ext_vector_type(16))) int   v16i;
typedef __attribute__((ext_vector_type(8)))  float v8f;

#define B_   8
#define D_   256
#define N_   4096
#define M_   4096

__device__ inline unsigned char f32_to_fp8_e4m3(float f) {
#if __has_builtin(__builtin_amdgcn_cvt_pk_fp8_f32)
  f = fminf(fmaxf(f, -448.f), 448.f);
  int r = __builtin_amdgcn_cvt_pk_fp8_f32(f, f, 0, false);
  return (unsigned char)(r & 0xFF);
#else
  unsigned u = __float_as_uint(f);
  unsigned char sign = (unsigned char)((u >> 24) & 0x80u);
  unsigned a = u & 0x7FFFFFFFu;
  if (a >= 0x7F800000u) return (unsigned char)(sign | 0x7Fu);
  if (a >= 0x43E80000u) return (unsigned char)(sign | 0x7Eu);
  int e = (int)(a >> 23) - 120;
  unsigned mant = a & 0x7FFFFFu;
  if (e >= 1) {
    unsigned m   = mant >> 20;
    unsigned rem = mant & 0xFFFFFu;
    if (rem > 0x80000u || (rem == 0x80000u && (m & 1u))) m++;
    if (m == 8u) { m = 0; e++; }
    if (e > 15 || (e == 15 && m == 7u)) return (unsigned char)(sign | 0x7Eu);
    return (unsigned char)(sign | (unsigned)(e << 3) | m);
  } else {
    int shift = 21 - e;
    if (shift > 31) return sign;
    unsigned sig  = 0x800000u | mant;
    unsigned m    = sig >> shift;
    unsigned rem  = sig & ((1u << shift) - 1u);
    unsigned half = 1u << (shift - 1);
    if (rem > half || (rem == half && (m & 1u))) m++;
    if (m >= 8u) return (unsigned char)(sign | 0x08u);
    return (unsigned char)(sign | m);
  }
#endif
}

__global__ void init_amax_kernel(unsigned* amax) {
  amax[0] = 0u;
  amax[1] = 0u;
}

__global__ __launch_bounds__(256)
void amax_kernel(const float* __restrict__ x, unsigned n, unsigned* __restrict__ out) {
  float m = 0.f;
  for (unsigned i = blockIdx.x * blockDim.x + threadIdx.x; i < n;
       i += gridDim.x * blockDim.x)
    m = fmaxf(m, fabsf(x[i]));
  __shared__ float red[256];
  red[threadIdx.x] = m;
  __syncthreads();
  for (int s = 128; s > 0; s >>= 1) {
    if ((int)threadIdx.x < s) red[threadIdx.x] = fmaxf(red[threadIdx.x], red[threadIdx.x + s]);
    __syncthreads();
  }
  if (threadIdx.x == 0) atomicMax(out, __float_as_uint(red[0]));
}

__global__ __launch_bounds__(256)
void quant_tr_kernel(const float* __restrict__ x, unsigned char* __restrict__ t,
                     const unsigned* __restrict__ amax_word) {
  __shared__ unsigned char lds[32][36];
  const int b  = blockIdx.z;
  const int d0 = blockIdx.y * 32;
  const int n0 = blockIdx.x * 32;
  const float amax  = fmaxf(__uint_as_float(*amax_word), 1e-12f);
  const float scale = 448.f / amax;

  const int nl  = threadIdx.x & 31;
  const int dl0 = threadIdx.x >> 5;
#pragma unroll
  for (int s = 0; s < 4; ++s) {
    int dl = dl0 + s * 8;
    float v = x[((size_t)b * D_ + (d0 + dl)) * N_ + n0 + nl];
    lds[nl][dl] = f32_to_fp8_e4m3(v * scale);
  }
  __syncthreads();
  const int nr = threadIdx.x >> 3;
  const int d4 = threadIdx.x & 7;
  unsigned w;
  __builtin_memcpy(&w, &lds[nr][d4 * 4], 4);
  *(unsigned*)(t + (((size_t)b * N_ + n0 + nr) << 8) + d0 + d4 * 4) = w;
}

union AFrag { v16i v; unsigned long long q[8]; };
union BFrag { v16i v; int4 x4[4]; };

__global__ __launch_bounds__(256, 2)
void gemm_fp8_kernel(const unsigned char* __restrict__ At,
                     const unsigned char* __restrict__ Bt,
                     const unsigned* __restrict__ amax,
                     float* __restrict__ out) {
  const int lane = threadIdx.x & 31;
  const int wave = threadIdx.x >> 5;
  const int wn   = wave >> 2;
  const int wm   = wave & 3;
  const int b    = blockIdx.z;
  const int n0   = blockIdx.y * 64  + wn * 32;
  const int m0   = blockIdx.x * 128 + wm * 32;
  const int hi   = lane >> 4;
  const int lr   = lane & 15;

  const unsigned char* Ab = At + (((size_t)b * N_ + n0 + lr) << 8);
  const unsigned char* Bb = Bt + (((size_t)b * M_ + m0 + lr) << 8);

  v8f acc[2][2] = {};

#pragma unroll
  for (int kk = 0; kk < D_; kk += 128) {
    AFrag a[2];
    BFrag bf[2];
#pragma unroll
    for (int i = 0; i < 2; ++i) {
      const unsigned char* p = Ab + ((size_t)(i * 16) << 8) + kk + hi * 16;
#pragma unroll
      for (int j = 0; j < 2; ++j)
#pragma unroll
        for (int tt = 0; tt < 4; ++tt)
          a[i].q[j * 4 + tt] = *(const unsigned long long*)(p + j * 64 + (tt >> 1) * 32 + (tt & 1) * 8);
    }
#pragma unroll
    for (int j = 0; j < 2; ++j) {
      const unsigned char* p = Bb + ((size_t)(j * 16) << 8) + kk + hi * 16;
#pragma unroll
      for (int g = 0; g < 4; ++g)
        bf[j].x4[g] = *(const int4*)(p + g * 32);
    }
#pragma unroll
    for (int i = 0; i < 2; ++i)
#pragma unroll
      for (int j = 0; j < 2; ++j)
        acc[i][j] = __builtin_amdgcn_wmma_f32_16x16x128_fp8_fp8(
            a[i].v, bf[j].v, (short)0, acc[i][j], false, false);
  }

  const float a0 = fmaxf(__uint_as_float(amax[0]), 1e-12f);
  const float a1 = fmaxf(__uint_as_float(amax[1]), 1e-12f);
  const float s  = a0 * a1 * (1.0f / 3211264.0f);

#pragma unroll
  for (int i = 0; i < 2; ++i)
#pragma unroll
    for (int j = 0; j < 2; ++j)
#pragma unroll
      for (int r = 0; r < 8; ++r) {
        const int row = n0 + i * 16 + r + hi * 8;
        const int col = m0 + j * 16 + lr;
        out[((size_t)b * N_ + row) * (size_t)M_ + col] = acc[i][j][r] * s;
      }
}

extern "C" void kernel_launch(void* const* d_in, const int* in_sizes, int n_in,
                              void* d_out, int out_size, void* d_ws, size_t ws_size,
                              hipStream_t stream) {
  const float* x0 = (const float*)d_in[0];
  const float* x1 = (const float*)d_in[1];
  float* out = (float*)d_out;

  unsigned char* ws   = (unsigned char*)d_ws;
  unsigned*      amax = (unsigned*)ws;
  unsigned char* At   = ws + 256;
  unsigned char* Bt   = At + (size_t)B_ * N_ * D_;

  const unsigned n_elems = (unsigned)B_ * D_ * N_;

  init_amax_kernel<<<1, 1, 0, stream>>>(amax);
  amax_kernel<<<2048, 256, 0, stream>>>(x0, n_elems, amax + 0);
  amax_kernel<<<2048, 256, 0, stream>>>(x1, n_elems, amax + 1);

  dim3 qgrid(N_ / 32, D_ / 32, B_);
  quant_tr_kernel<<<qgrid, 256, 0, stream>>>(x0, At, amax + 0);
  quant_tr_kernel<<<qgrid, 256, 0, stream>>>(x1, Bt, amax + 1);

  dim3 ggrid(M_ / 128, N_ / 64, B_);
  gemm_fp8_kernel<<<ggrid, 256, 0, stream>>>(At, Bt, amax, out);
}
